// GCNLayer_V2_52999896432940
// MI455X (gfx1250) — matrix operand order repaired and checked by reading, not compiled, not run on hardware
//
#include <hip/hip_runtime.h>
#include <hip/hip_bf16.h>
#include <math.h>

// GCN layer with trimmed-mean aggregation, MI455X (gfx1250, wave32).
//
// Kernel 1 (aggregation): L2-resident random gather (h = 10MB << 192MB L2),
//   O(D^2) branch-free stable ranking per feature column, thread-per-column.
// Kernel 2 (linear): D = relu(accum @ W + bias) via V_WMMA_F32_16X16X4_F32
//   (full f32 precision), fallback to V_WMMA_F32_16X16X32_F16 if the f32
//   builtin is unavailable in this toolchain.

#define N_NODES  20000
#define MAX_DEG  32
#define FDIM     128
#define TRIMF    0.45f
#define NTHRESH  3

typedef __attribute__((ext_vector_type(2)))  float    v2f;
typedef __attribute__((ext_vector_type(8)))  float    v8f;
typedef __attribute__((ext_vector_type(16))) _Float16 v16h;

// ---------------------------------------------------------------------------
// Kernel 1: trimmed-mean neighborhood aggregation.
// grid = N_NODES blocks, block = 128 threads (one thread per feature column,
// 4 waves). Neighbor tile [32][128] staged in LDS: address = j*128 + f, so a
// wave32 (f contiguous) hits 32 distinct banks -> conflict-free.
// ---------------------------------------------------------------------------
__global__ __launch_bounds__(FDIM) void gcn_aggregate(
    const float* __restrict__ h, const float* __restrict__ nrm,
    const int* __restrict__ nbr, const int* __restrict__ deg,
    float* __restrict__ accum)
{
  __shared__ float sH[MAX_DEG * FDIM];   // neighbor features (keys)
  __shared__ float sNrm[MAX_DEG];        // neighbor norms
  __shared__ int   sIdx[MAX_DEG];        // neighbor ids

  const int i = blockIdx.x;
  const int f = threadIdx.x;
  const int n = deg[i];                  // uniform -> scalar load
  const float ni = nrm[i];

  if (f < MAX_DEG) {
    int nj = (f < n) ? nbr[(size_t)i * MAX_DEG + f] : 0;
    sIdx[f] = nj;
    sNrm[f] = nrm[nj];
  }
  __syncthreads();

  // Stage valid neighbor rows: coalesced 512B row loads, L2 hits.
  for (int j = 0; j < n; ++j) {
    sH[j * FDIM + f] = h[(size_t)sIdx[j] * FDIM + f];
  }
  __syncthreads();

  const float self0 = h[(size_t)i * FDIM + f] * ni;
  float acc;
  if (n <= NTHRESH) {
    acc = ((float)n * self0) * ni;       // small-degree branch
  } else {
    // trim count b, exactly as reference (f32 floor for the cap)
    int b_raw = n / 2 - (1 - (n & 1));
    int b_cap = (int)floorf((float)n * TRIMF);
    int b = (b_raw < b_cap) ? b_raw : b_cap;
    if (b < 1) b = 1;
    const int hi = n - b;

    // Stable rank-by-counting == stable argsort selection (ties broken by
    // original slot order, matching jnp.argsort).
    float sum = 0.0f;
    for (int j = 0; j < n; ++j) {
      const float kj = sH[j * FDIM + f];
      int rank = 0;
#pragma unroll 4
      for (int k = 0; k < n; ++k) {
        const float kk = sH[k * FDIM + f];
        rank += (int)(kk < kj) | ((int)(kk == kj) & (int)(k < j));
      }
      if (rank >= b && rank < hi) sum += kj * sNrm[j];  // msg = key * src-norm
    }
    acc = (sum + self0 * (float)(2 * b)) * ni;          // trim-compensated
  }
  accum[(size_t)i * FDIM + f] = acc;
}

// ---------------------------------------------------------------------------
// Kernel 2: out = relu(accum @ W + bias) via WMMA.
// grid = 1250 blocks (16-row strips), block = 256 threads = 8 waves, each
// wave owns one 16-wide output column tile. A strip staged in LDS (f32).
// ---------------------------------------------------------------------------
__global__ __launch_bounds__(256) void gcn_linear(
    const float* __restrict__ accum, const float* __restrict__ W,
    const float* __restrict__ bias, float* __restrict__ out)
{
#if defined(__gfx1250__)
  __shared__ float sA[16 * FDIM];        // 16x128 accum strip, 8KB

  const int row0 = blockIdx.x * 16;
  const int tid  = threadIdx.x;

  for (int t = tid; t < 16 * FDIM; t += 256) {
    sA[t] = accum[(size_t)row0 * FDIM + t];
  }
  __syncthreads();

  const int lane = tid & 31;
  const int half = lane >> 4;            // lane-half selects K sub-slice
  const int l16  = lane & 15;
  const int col0 = (tid >> 5) * 16;      // wave id -> column tile

  v8f c = {};                            // C/D: C[v + 8*half][l16]

#if __has_builtin(__builtin_amdgcn_wmma_f32_16x16x4_f32)
  // Pure-f32 WMMA: A 16x4 (VGPR0: K=0/2 per half, VGPR1: K=1/3),
  //                B 4x16 mirrored. 32 WMMAs accumulate K=128.
  for (int k0 = 0; k0 < FDIM; k0 += 4) {
    const int ka = k0 + 2 * half;
    v2f a, b;
    a.x = sA[l16 * FDIM + ka];
    a.y = sA[l16 * FDIM + ka + 1];
    b.x = W[(size_t)ka * FDIM + col0 + l16];
    b.y = W[(size_t)(ka + 1) * FDIM + col0 + l16];
    c = __builtin_amdgcn_wmma_f32_16x16x4_f32(
        false, a, false, b, (short)0, c, false, false);
  }
#else
  // Fallback: codegen-confirmed f16 WMMA, f32 accumulate (K=32 per op).
  for (int k0 = 0; k0 < FDIM; k0 += 32) {
    v16h a, b;
    const int kbA = k0 + half * 16;      // A: K = kbA..kbA+15
#pragma unroll
    for (int e = 0; e < 8; ++e) {
      a[e]     = (_Float16)sA[l16 * FDIM + kbA + e];
      a[e + 8] = (_Float16)sA[l16 * FDIM + kbA + 8 + e];
    }
    const int kbB = k0 + half * 16;      // B: K = kbB + 2r, 2r+1 per VGPR r
#pragma unroll
    for (int r = 0; r < 8; ++r) {
      b[2 * r]     = (_Float16)W[(size_t)(kbB + 2 * r) * FDIM + col0 + l16];
      b[2 * r + 1] = (_Float16)W[(size_t)(kbB + 2 * r + 1) * FDIM + col0 + l16];
    }
    c = __builtin_amdgcn_wmma_f32_16x16x32_f16(
        false, a, false, b, (short)0, c, false, false);
  }
#endif

  // Epilogue: bias + ReLU, fused into the C/D layout store.
  const float bv = bias[col0 + l16];
#pragma unroll
  for (int v = 0; v < 8; ++v) {
    const int m = half * 8 + v;
    const float o = c[v] + bv;
    out[(size_t)(row0 + m) * FDIM + col0 + l16] = fmaxf(o, 0.0f);
  }
#endif  // __gfx1250__
}

// ---------------------------------------------------------------------------
extern "C" void kernel_launch(void* const* d_in, const int* in_sizes, int n_in,
                              void* d_out, int out_size, void* d_ws, size_t ws_size,
                              hipStream_t stream) {
  const float* h    = (const float*)d_in[0];   // [N, 128]
  const float* nrm  = (const float*)d_in[1];   // [N, 1]
  const float* W    = (const float*)d_in[2];   // [128, 128]
  const float* bias = (const float*)d_in[3];   // [128]
  const int*   nbr  = (const int*)d_in[4];     // [N, 32]
  const int*   deg  = (const int*)d_in[5];     // [N]
  float* out   = (float*)d_out;                // [N, 128]
  float* accum = (float*)d_ws;                 // scratch [N, 128] f32 = 10.24 MB

  gcn_aggregate<<<N_NODES, FDIM, 0, stream>>>(h, nrm, nbr, deg, accum);
  gcn_linear<<<N_NODES / 16, 256, 0, stream>>>(accum, W, bias, out);
}
